// RAJNIAttention_59992103191164
// MI455X (gfx1250) — matrix operand order repaired and checked by reading, not compiled, not run on hardware
//
#include <hip/hip_runtime.h>
#include <math.h>

typedef __attribute__((ext_vector_type(16))) __bf16 v16bf;
typedef __attribute__((ext_vector_type(8)))  __bf16 v8bf;
typedef __attribute__((ext_vector_type(8)))  float  v8f;

#define BB 32
#define NN 577
#define CC 768
#define HH 12
#define DD 64
#define C3 2304
#define NPAD 608            // padded token count: multiple of 32 (19 key tiles, 38 q tiles)
#define MROWS (BB*NN)       // 18464 (divisible by 16)
#define QTILES (NPAD/16)    // 38
#define KTILES32 (NPAD/32)  // 19
#define SCALE_ATT 0.125f

union V16 { v16bf v; v8bf h2[2]; unsigned short s[16]; };
union V8F { v8f v; float f[8]; };

__device__ __forceinline__ unsigned short bf16_bits(float x) {
  union { float f; unsigned u; } c; c.f = x;
  unsigned r = c.u + 0x7FFFu + ((c.u >> 16) & 1u);   // RNE
  return (unsigned short)(r >> 16);
}

__device__ __forceinline__ v8f wmma_bf16(v16bf a, v16bf b, v8f c) {
  return __builtin_amdgcn_wmma_f32_16x16x32_bf16(false, a, false, b, (short)0, c,
                                                 false, false);
}

// ---------------------------------------------------------------- TDM support
#if defined(__has_builtin)
#if __has_builtin(__builtin_amdgcn_tensor_load_to_lds)
#define HAVE_TDM 1
#endif
#endif

#ifdef HAVE_TDM
typedef unsigned int __attribute__((ext_vector_type(4))) u32x4;
typedef int          __attribute__((ext_vector_type(4))) i32x4;
typedef int          __attribute__((ext_vector_type(8))) i32x8;

// 2D tile load, bf16 elements (data_size = 2B).
// gptr points at tile start; tensor width/stride given in elements.
__device__ __forceinline__ void tdm_load_2d(unsigned lds_off, const void* gptr,
                                            unsigned tensor_w, unsigned tensor_h,
                                            unsigned tile_w, unsigned tile_h,
                                            unsigned row_stride) {
  unsigned long long ga = (unsigned long long)gptr;
  u32x4 g0;
  g0[0] = 1u;                                             // count=1, user mode
  g0[1] = lds_off;                                        // lds_addr (bytes)
  g0[2] = (unsigned)(ga & 0xFFFFFFFFu);                   // global_addr[31:0]
  g0[3] = (unsigned)((ga >> 32) & 0x01FFFFFFu) | (2u << 30);  // addr[56:32] | type=2
  i32x8 g1;
  g1[0] = 0x00010000;                                     // data_size=1 (2 bytes)
  g1[1] = (int)((tensor_w & 0xFFFFu) << 16);              // tensor_dim0[15:0]
  g1[2] = (int)(((tensor_w >> 16) & 0xFFFFu) | ((tensor_h & 0xFFFFu) << 16));
  g1[3] = (int)(((tensor_h >> 16) & 0xFFFFu) | ((tile_w & 0xFFFFu) << 16));
  g1[4] = (int)(tile_h & 0xFFFFu);                        // tile_dim1 (tile_dim2=0)
  g1[5] = (int)row_stride;                                // tensor_dim0_stride[31:0]
  g1[6] = 0;
  g1[7] = 0;
  i32x4 z4 = {0, 0, 0, 0};
#if __clang_major__ >= 23
  i32x8 z8 = {0, 0, 0, 0, 0, 0, 0, 0};
  __builtin_amdgcn_tensor_load_to_lds(g0, g1, z4, z4, z8, 0);
#else
  __builtin_amdgcn_tensor_load_to_lds(g0, g1, z4, z4, 0);
#endif
}
#endif

// ---------------------------------------------------------------- fp32 -> bf16
__global__ void k_f2bf(const float* __restrict__ in, unsigned short* __restrict__ out,
                       int n) {
  int i = blockIdx.x * blockDim.x + threadIdx.x;
  if (i < n) out[i] = bf16_bits(in[i]);
}

// ---------------------------------------------------------------- weight pack
// Pack W[K,N] fp32 into bf16 WMMA B-fragment order:
// P[(((size_t)nt*(K/32) + kt)*32 + lane)*16 + e] = W[(kt*32 + half*16 + e)*N + nt*16 + c16]
__global__ void k_packW(const float* __restrict__ W, unsigned short* __restrict__ P,
                        int K, int N) {
  size_t i = (size_t)blockIdx.x * blockDim.x + threadIdx.x;
  size_t total = (size_t)K * N;
  if (i >= total) return;
  int e = (int)(i & 15);
  size_t j = i >> 4;
  int lane = (int)(j & 31); j >>= 5;
  int kt = (int)(j % (size_t)(K / 32));
  int nt = (int)(j / (size_t)(K / 32));
  int half = lane >> 4, c16 = lane & 15;
  P[i] = bf16_bits(W[(size_t)(kt * 32 + half * 16 + e) * N + nt * 16 + c16]);
}

// ---------------------------------------------------------------- qkv = x @ w_qkv + b
// A: [MROWS, CC] bf16 row-major; Bp: packed; out fp32 [MROWS, C3].
// One wave computes a 16x64 tile (A fragment reused across 4 WMMAs).
__global__ void k_qkv_gemm(const unsigned short* __restrict__ A,
                           const unsigned short* __restrict__ Bp,
                           const float* __restrict__ bias,
                           float* __restrict__ out) {
  const int n4tiles = C3 / 64;                   // 36
  const int total   = (MROWS / 16) * n4tiles;    // 41544
  int wid = blockIdx.x * (blockDim.x >> 5) + (threadIdx.x >> 5);
  if (wid >= total) return;
  int mt = wid / n4tiles, n4 = wid % n4tiles;
  int lane = threadIdx.x & 31, half = lane >> 4, c16 = lane & 15;
  const unsigned short* arow = A + (size_t)(mt * 16 + c16) * CC;

  V8F acc[4];
  #pragma unroll
  for (int q = 0; q < 4; q++)
    #pragma unroll
    for (int r = 0; r < 8; r++) acc[q].f[r] = 0.f;

  for (int kt = 0; kt < CC / 32; kt++) {
    int kb = kt * 32;
    __builtin_prefetch(arow + kb + 128, 0, 1);   // global_prefetch_b8
    V16 ua;   // A 16x32 bf16 layout
    ua.h2[0] = *reinterpret_cast<const v8bf*>(arow + kb + half * 16);
    ua.h2[1] = *reinterpret_cast<const v8bf*>(arow + kb + half * 16 + 8);
    #pragma unroll
    for (int q = 0; q < 4; q++) {
      int nt = n4 * 4 + q;
      v16bf ub = *reinterpret_cast<const v16bf*>(
          Bp + (((size_t)nt * (CC / 32) + kt) * 32 + lane) * 16);
      acc[q].v = wmma_bf16(ua.v, ub, acc[q].v);
    }
  }
  #pragma unroll
  for (int q = 0; q < 4; q++) {
    int col = (n4 * 4 + q) * 16 + c16;
    float bcol = bias[col];
    #pragma unroll
    for (int r = 0; r < 8; r++) {                // C layout: row = r + 8*half
      int row = mt * 16 + r + 8 * half;
      out[(size_t)row * C3 + col] = acc[q].f[r] + bcol;
    }
  }
}

// ---------------------------------------------------------------- q-norm importance
__global__ void k_importance(const float* __restrict__ qkv, float* __restrict__ scores) {
  int wid = blockIdx.x * (blockDim.x >> 5) + (threadIdx.x >> 5);
  if (wid >= MROWS) return;
  int lane = threadIdx.x & 31;
  const float* q = qkv + (size_t)wid * C3;
  float acc = 0.f;
  #pragma unroll
  for (int h = 0; h < HH; h++) {
    float v0 = q[h * DD + lane], v1 = q[h * DD + 32 + lane];
    float s = v0 * v0 + v1 * v1;
    #pragma unroll
    for (int m = 16; m > 0; m >>= 1) s += __shfl_xor(s, m, 32);
    acc += sqrtf(s);
  }
  if (lane == 0) scores[wid] = acc / (float)HH;
}

// ---------------------------------------------------------------- sort + quantile
__global__ void k_sort_quantile(const float* __restrict__ scores,
                                float* __restrict__ sorted_ls,
                                float* __restrict__ Dl) {
  int b = blockIdx.x, t = threadIdx.x;           // 512 threads
  __shared__ float ls[1024];
  __shared__ float red[512];
  for (int i = t; i < 1024; i += 512)
    ls[i] = (i < (NN - 1)) ? logf(scores[b * NN + 1 + i] + 1e-12f)
                           : __builtin_inff();
  __syncthreads();
  for (int k = 2; k <= 1024; k <<= 1) {
    for (int j = k >> 1; j > 0; j >>= 1) {
      for (int i = t; i < 1024; i += 512) {
        int ixj = i ^ j;
        if (ixj > i) {
          bool up = ((i & k) == 0);
          float a = ls[i], c = ls[ixj];
          if ((a > c) == up) { ls[i] = c; ls[ixj] = a; }
        }
      }
      __syncthreads();
    }
  }
  for (int i = t; i < (NN - 1); i += 512) sorted_ls[b * (NN - 1) + i] = ls[i];
  float qtl = ls[431] + 0.25f * (ls[432] - ls[431]);   // quantile(0.75), linear
  float s = 0.f;
  for (int i = t; i < (NN - 1); i += 512) {
    float d = qtl - ls[i];
    s += (d > 0.f) ? d : 0.f;
  }
  red[t] = s; __syncthreads();
  for (int o = 256; o > 0; o >>= 1) { if (t < o) red[t] += red[t + o]; __syncthreads(); }
  if (t == 0) Dl[b] = (red[0] / (float)(NN - 1)) / fabsf(qtl);
}

// ---------------------------------------------------------------- keep count
__global__ void k_keep(const float* __restrict__ Dl, int* __restrict__ keepbuf) {
  int t = threadIdx.x;                            // 32
  float v = Dl[t];
  #pragma unroll
  for (int m = 16; m > 0; m >>= 1) v += __shfl_xor(v, m, 32);
  if (t == 0) {
    float D = v / (float)BB;
    float kr = fmaxf(0.6f, expf(-2.5f * D));
    int keep = (int)(kr * (float)(NN - 1));
    if (keep < 1) keep = 1;
    if (keep > NN - 1) keep = NN - 1;
    keepbuf[0] = keep;
    keepbuf[1] = keep + 1;                        // Np
  }
}

// ---------------------------------------------------------------- top-k -> sorted keep_idx
__global__ void k_topk(const float* __restrict__ scores,
                       const float* __restrict__ sorted_ls,
                       const int* __restrict__ keepbuf,
                       int* __restrict__ keepidx) {
  int b = blockIdx.x, t = threadIdx.x;            // 1024 threads
  int keep = keepbuf[0];
  float thr = sorted_ls[b * (NN - 1) + (NN - 1 - keep)];
  __shared__ int gg[1024]; __shared__ int eq[1024]; __shared__ int kp[1024];
  float ls = (t < NN - 1) ? logf(scores[b * NN + 1 + t] + 1e-12f) : -__builtin_inff();
  int g = (t < NN - 1) && (ls > thr);
  int e = (t < NN - 1) && (ls == thr);
  gg[t] = g; eq[t] = e; __syncthreads();
  for (int o = 1; o < 1024; o <<= 1) {
    int vg = (t >= o) ? gg[t - o] : 0;
    int ve = (t >= o) ? eq[t - o] : 0;
    __syncthreads();
    gg[t] += vg; eq[t] += ve;
    __syncthreads();
  }
  int Gtot = gg[1023];
  int kept = g || (e && ((eq[t] - e) < (keep - Gtot)));
  kp[t] = kept; __syncthreads();
  for (int o = 1; o < 1024; o <<= 1) {
    int vk = (t >= o) ? kp[t - o] : 0;
    __syncthreads();
    kp[t] += vk;
    __syncthreads();
  }
  if (kept) keepidx[b * NN + (kp[t] - kept) + 1] = t + 1;
  if (t == 0) keepidx[b * NN] = 0;
}

// ---------------------------------------------------------------- gather to bf16
// qh/kk: [b*HH+h][NPAD][DD], vt: [b*HH+h][DD][NPAD]
__global__ void k_gather(const float* __restrict__ qkv,
                         const int* __restrict__ keepidx,
                         const int* __restrict__ keepbuf,
                         unsigned short* __restrict__ qh,
                         unsigned short* __restrict__ kk,
                         unsigned short* __restrict__ vt) {
  int i = blockIdx.x % NPAD, b = blockIdx.x / NPAD;
  int Np = keepbuf[1];
  int t = threadIdx.x;                            // 256
  bool valid = (i < Np);
  int src = valid ? keepidx[b * NN + i] : 0;
  const float* p = qkv + ((size_t)b * NN + src) * C3;
  for (int c = t; c < C3; c += 256) {
    int part = c / CC, rem = c - part * CC, h = rem >> 6, d = rem & 63;
    unsigned short v = valid ? bf16_bits(p[c]) : (unsigned short)0;
    size_t bh = (size_t)(b * HH + h);
    if (part == 0)      qh[(bh * NPAD + i) * DD + d] = v;
    else if (part == 1) kk[(bh * NPAD + i) * DD + d] = v;
    else                vt[(bh * DD + d) * NPAD + i] = v;
  }
}

// ---------------------------------------------------------------- flash attention
// one wave per (b, h, 16-query tile). K/V tiles staged to LDS via TDM when available.
__global__ void k_attn(const unsigned short* __restrict__ qh,
                       const unsigned short* __restrict__ kk,
                       const unsigned short* __restrict__ vt,
                       const int* __restrict__ keepbuf,
                       unsigned short* __restrict__ attnb) {
  int qt = blockIdx.x % QTILES;
  int bh = blockIdx.x / QTILES;                   // 0..383
  int b = bh / HH, h = bh % HH;
  int Np = keepbuf[1];
  if (qt * 16 >= Np) return;
  int lane = threadIdx.x & 31, half = lane >> 4, c16 = lane & 15;

  __shared__ unsigned short pld[16 * 32] __attribute__((aligned(64)));
#ifdef HAVE_TDM
  __shared__ unsigned short kbuf[32 * DD] __attribute__((aligned(64)));  // [key][d]
  __shared__ unsigned short vbuf[DD * 32] __attribute__((aligned(64)));  // [d][key]
  unsigned kbuf_off = (unsigned)(size_t)&kbuf[0];
  unsigned vbuf_off = (unsigned)(size_t)&vbuf[0];
#endif

  const unsigned short* qbase = qh + ((size_t)bh * NPAD + qt * 16 + c16) * DD;
  V16 aq[2];
  #pragma unroll
  for (int kc = 0; kc < 2; kc++) {
    aq[kc].h2[0] = *reinterpret_cast<const v8bf*>(qbase + kc * 32 + half * 16);
    aq[kc].h2[1] = *reinterpret_cast<const v8bf*>(qbase + kc * 32 + half * 16 + 8);
  }

  float m[8], l[8];
  V8F O[4];
  #pragma unroll
  for (int r = 0; r < 8; r++) { m[r] = -__builtin_inff(); l[r] = 0.f; }
  #pragma unroll
  for (int nt = 0; nt < 4; nt++)
    #pragma unroll
    for (int r = 0; r < 8; r++) O[nt].f[r] = 0.f;

  int ktiles = (Np + 31) / 32;
  for (int kt = 0; kt < ktiles; kt++) {
#ifdef HAVE_TDM
    // WAR guard: previous iteration's LDS readers must drain before TDM rewrites
    asm volatile("s_wait_dscnt 0x0" ::: "memory");
    // K tile: 32 keys x 64 d, row stride 64 (contiguous panel)
    tdm_load_2d(kbuf_off, kk + ((size_t)bh * NPAD + kt * 32) * DD,
                DD, 32, DD, 32, DD);
    // V tile: 64 d-rows x 32 keys, row stride NPAD (strided 2D tile)
    tdm_load_2d(vbuf_off, vt + (size_t)bh * DD * NPAD + kt * 32,
                NPAD, DD, 32, DD, NPAD);
    __builtin_amdgcn_s_wait_tensorcnt(0);
#endif
    // S = Q K^T for 32 keys -> two 16x16 accumulators
    V8F S[2];
    #pragma unroll
    for (int s = 0; s < 2; s++) {
#ifdef HAVE_TDM
      const unsigned short* kb2 = kbuf + (s * 16 + c16) * DD;
#else
      const unsigned short* kb2 =
          kk + ((size_t)bh * NPAD + kt * 32 + s * 16 + c16) * DD;
#endif
      V8F acc;
      #pragma unroll
      for (int r = 0; r < 8; r++) acc.f[r] = 0.f;
      #pragma unroll
      for (int kc = 0; kc < 2; kc++) {
        v16bf bk = *reinterpret_cast<const v16bf*>(kb2 + kc * 32 + half * 16);
        acc.v = wmma_bf16(aq[kc].v, bk, acc.v);
      }
      S[s] = acc;
    }
    // online softmax (row = r + 8*half)
    float scl[8];
    #pragma unroll
    for (int r = 0; r < 8; r++) {
      float s0 = S[0].f[r] * SCALE_ATT, s1 = S[1].f[r] * SCALE_ATT;
      int key0 = kt * 32 + c16, key1 = key0 + 16;
      if (key0 >= Np) s0 = -__builtin_inff();
      if (key1 >= Np) s1 = -__builtin_inff();
      float mx = fmaxf(s0, s1);
      #pragma unroll
      for (int o = 1; o < 16; o <<= 1) mx = fmaxf(mx, __shfl_xor(mx, o, 32));
      float mnew = fmaxf(m[r], mx);
      float sc = expf(m[r] - mnew);
      float e0 = (key0 < Np) ? expf(s0 - mnew) : 0.f;
      float e1 = (key1 < Np) ? expf(s1 - mnew) : 0.f;
      float rs = e0 + e1;
      #pragma unroll
      for (int o = 1; o < 16; o <<= 1) rs += __shfl_xor(rs, o, 32);
      l[r] = l[r] * sc + rs;
      m[r] = mnew;
      scl[r] = sc;
      pld[(r + 8 * half) * 32 + c16]      = bf16_bits(e0);
      pld[(r + 8 * half) * 32 + 16 + c16] = bf16_bits(e1);
    }
    #pragma unroll
    for (int nt = 0; nt < 4; nt++)
      #pragma unroll
      for (int r = 0; r < 8; r++) O[nt].f[r] *= scl[r];
    __syncthreads();
    // P (16x32) as A-fragment from LDS; V B-fragments (LDS via TDM, or global)
    V16 ap;
    ap.h2[0] = *reinterpret_cast<const v8bf*>(&pld[c16 * 32 + half * 16]);
    ap.h2[1] = *reinterpret_cast<const v8bf*>(&pld[c16 * 32 + half * 16 + 8]);
    #pragma unroll
    for (int nt = 0; nt < 4; nt++) {
#ifdef HAVE_TDM
      const unsigned short* vb = vbuf + (nt * 16 + c16) * 32 + half * 16;
#else
      const unsigned short* vb =
          vt + ((size_t)bh * DD + nt * 16 + c16) * NPAD + kt * 32 + half * 16;
#endif
      v16bf bv = *reinterpret_cast<const v16bf*>(vb);
      O[nt].v = wmma_bf16(ap.v, bv, O[nt].v);
    }
    __syncthreads();
  }
  #pragma unroll
  for (int r = 0; r < 8; r++) {
    int row = qt * 16 + r + 8 * half;
    if (row < Np) {
      float inv = 1.f / fmaxf(l[r], 1e-20f);
      #pragma unroll
      for (int nt = 0; nt < 4; nt++)
        attnb[((size_t)b * NPAD + row) * CC + h * DD + nt * 16 + c16] =
            bf16_bits(O[nt].f[r] * inv);
    }
  }
}

// ---------------------------------------------------------------- out = attn @ w_proj + b
__global__ void k_proj_gemm(const unsigned short* __restrict__ A,
                            const unsigned short* __restrict__ Bp,
                            const float* __restrict__ bias,
                            const int* __restrict__ keepbuf,
                            float* __restrict__ out) {
  const int n4tiles = CC / 64;                    // 12
  const int mtiles  = QTILES;                     // 38
  int wid = blockIdx.x * (blockDim.x >> 5) + (threadIdx.x >> 5);
  int total = BB * mtiles * n4tiles;
  if (wid >= total) return;
  int b  = wid / (mtiles * n4tiles);
  int rr = wid % (mtiles * n4tiles);
  int mt = rr / n4tiles, n4 = rr % n4tiles;
  int Np = keepbuf[1];
  if (mt * 16 >= Np) return;
  int lane = threadIdx.x & 31, half = lane >> 4, c16 = lane & 15;
  const unsigned short* arow = A + ((size_t)b * NPAD + mt * 16 + c16) * CC;

  V8F acc[4];
  #pragma unroll
  for (int q = 0; q < 4; q++)
    #pragma unroll
    for (int r = 0; r < 8; r++) acc[q].f[r] = 0.f;

  for (int kt = 0; kt < CC / 32; kt++) {
    int kb = kt * 32;
    __builtin_prefetch(arow + kb + 128, 0, 1);
    V16 ua;
    ua.h2[0] = *reinterpret_cast<const v8bf*>(arow + kb + half * 16);
    ua.h2[1] = *reinterpret_cast<const v8bf*>(arow + kb + half * 16 + 8);
    #pragma unroll
    for (int q = 0; q < 4; q++) {
      int nt = n4 * 4 + q;
      v16bf ub = *reinterpret_cast<const v16bf*>(
          Bp + (((size_t)nt * (CC / 32) + kt) * 32 + lane) * 16);
      acc[q].v = wmma_bf16(ua.v, ub, acc[q].v);
    }
  }
  #pragma unroll
  for (int q = 0; q < 4; q++) {
    int col = (n4 * 4 + q) * 16 + c16;
    float bcol = bias[col];
    #pragma unroll
    for (int r = 0; r < 8; r++) {
      int row = mt * 16 + r + 8 * half;
      if (row < Np)
        out[((size_t)b * Np + row) * CC + col] = acc[q].f[r] + bcol;
    }
  }
}

// ---------------------------------------------------------------- keep_idx tail of d_out
__global__ void k_write_idx(const int* __restrict__ keepidx,
                            const int* __restrict__ keepbuf,
                            float* __restrict__ out) {
  int i = blockIdx.x * blockDim.x + threadIdx.x;
  if (i >= BB * NN) return;
  int Np = keepbuf[1];
  int b = i / NN, t = i % NN;
  if (t < Np)
    ((int*)out)[(size_t)BB * Np * CC + (size_t)b * Np + t] = keepidx[b * NN + t];
}

// ================================================================ launch
extern "C" void kernel_launch(void* const* d_in, const int* in_sizes, int n_in,
                              void* d_out, int out_size, void* d_ws, size_t ws_size,
                              hipStream_t stream) {
  (void)in_sizes; (void)n_in; (void)out_size; (void)ws_size;
  const float* x      = (const float*)d_in[0];
  const float* w_qkv  = (const float*)d_in[1];
  const float* b_qkv  = (const float*)d_in[2];
  const float* w_proj = (const float*)d_in[3];
  const float* b_proj = (const float*)d_in[4];
  float* out = (float*)d_out;

  char* w = (char*)d_ws;
  size_t off = 0;
  auto take = [&](size_t bytes) -> size_t {
    size_t o = off; off += (bytes + 255) & ~(size_t)255; return o;
  };
  size_t o_qkv    = take((size_t)MROWS * C3 * 4);
  size_t o_xb     = take((size_t)MROWS * CC * 2);
  size_t o_wqkvp  = take((size_t)CC * C3 * 2);
  size_t o_wprojp = take((size_t)CC * CC * 2);
  size_t o_scores = take((size_t)MROWS * 4);
  size_t o_sorted = take((size_t)BB * (NN - 1) * 4);
  size_t o_Dl     = take((size_t)BB * 4);
  size_t o_keep   = take(64);
  size_t o_kidx   = take((size_t)BB * NN * 4);
  size_t o_qh     = take((size_t)BB * HH * NPAD * DD * 2);
  size_t o_kk     = take((size_t)BB * HH * NPAD * DD * 2);
  size_t o_vt     = take((size_t)BB * HH * NPAD * DD * 2);
  size_t o_attnb  = take((size_t)BB * NPAD * CC * 2);

  float*          qkv    = (float*)(w + o_qkv);
  unsigned short* xb     = (unsigned short*)(w + o_xb);
  unsigned short* wqkvp  = (unsigned short*)(w + o_wqkvp);
  unsigned short* wprojp = (unsigned short*)(w + o_wprojp);
  float*          scores = (float*)(w + o_scores);
  float*          sorted = (float*)(w + o_sorted);
  float*          Dl     = (float*)(w + o_Dl);
  int*            keepb  = (int*)(w + o_keep);
  int*            kidx   = (int*)(w + o_kidx);
  unsigned short* qh     = (unsigned short*)(w + o_qh);
  unsigned short* kk     = (unsigned short*)(w + o_kk);
  unsigned short* vt     = (unsigned short*)(w + o_vt);
  unsigned short* attnb  = (unsigned short*)(w + o_attnb);

  const int n_x = MROWS * CC;

  k_f2bf<<<(n_x + 255) / 256, 256, 0, stream>>>(x, xb, n_x);
  k_packW<<<(CC * C3 + 255) / 256, 256, 0, stream>>>(w_qkv, wqkvp, CC, C3);
  k_packW<<<(CC * CC + 255) / 256, 256, 0, stream>>>(w_proj, wprojp, CC, CC);

  {
    int waves = (MROWS / 16) * (C3 / 64);
    k_qkv_gemm<<<(waves + 7) / 8, 256, 0, stream>>>(xb, wqkvp, b_qkv, qkv);
  }
  k_importance<<<(MROWS + 7) / 8, 256, 0, stream>>>(qkv, scores);
  k_sort_quantile<<<BB, 512, 0, stream>>>(scores, sorted, Dl);
  k_keep<<<1, 32, 0, stream>>>(Dl, keepb);
  k_topk<<<BB, 1024, 0, stream>>>(scores, sorted, keepb, kidx);
  k_gather<<<BB * NPAD, 256, 0, stream>>>(qkv, kidx, keepb, qh, kk, vt);
  k_attn<<<BB * HH * QTILES, 32, 0, stream>>>(qh, kk, vt, keepb, attnb);
  {
    int waves = BB * QTILES * (CC / 64);
    k_proj_gemm<<<(waves + 7) / 8, 256, 0, stream>>>(attnb, wprojp, b_proj, keepb, out);
  }
  k_write_idx<<<(BB * NN + 255) / 256, 256, 0, stream>>>(kidx, keepb, out);
}
